// CMSAModel_82583631167636
// MI455X (gfx1250) — compile-verified, matrix operand order checked by reading, not run on hardware
//
#include <hip/hip_runtime.h>
#include <hip/hip_bf16.h>
#include <stdint.h>

typedef __bf16 bf16_t;
typedef __attribute__((ext_vector_type(16))) __bf16 v16bf;
typedef __attribute__((ext_vector_type(8)))  float   v8f;
typedef __attribute__((ext_vector_type(4)))  unsigned int v4u;
typedef __attribute__((ext_vector_type(8)))  unsigned int v8u;

#define BATCH 4
#define SEQ   4096
#define DIM   256
#define CPAD  544   // 520 padded to 17*32

typedef __attribute__((address_space(3))) void lds_void_t;

// ---- fragment helpers ------------------------------------------------------
// bf16 16x16x32 WMMA A/B fragment: lanes 0-15 hold K {0..7,16..23}, lanes
// 16-31 hold K {8..15,24..31} (row/col = lane%16). From a row-major 32-element
// K chunk that is two contiguous 16B loads at +koff and +koff+16 elements,
// koff = (lane<16)?0:8.

struct Frag32B { uint4 lo, hi; };

static __device__ __forceinline__ v16bf make_frag(uint4 lo, uint4 hi) {
  Frag32B f{lo, hi};
  return __builtin_bit_cast(v16bf, f);
}

static __device__ __forceinline__ v16bf load_frag(const bf16_t* p) {
  return make_frag(*(const uint4*)p, *(const uint4*)(p + 16));
}

// ---- Tensor Data Mover (inline asm; portable across toolchains) ------------
// 2-D D# descriptor per CDNA5 ISA sections 8.3/8.4: load a (rows x elems_x)
// bf16 tile whose rows are row_stride elements apart into contiguous LDS at
// lds_off.  Groups 2/3 omitted (NULL) -> tensor up to 2D.
static __device__ __forceinline__ void tdm_load_2d(uint32_t lds_off,
                                                   const void* gptr,
                                                   uint32_t elems_x,
                                                   uint32_t rows,
                                                   uint32_t row_stride,
                                                   uint32_t tensor_rows) {
  unsigned long long ga = (unsigned long long)(uintptr_t)gptr;
  v4u g0;
  g0[0] = 1u;                                         // count=1, user mode
  g0[1] = lds_off;                                    // lds_addr (bytes)
  g0[2] = (unsigned int)ga;                           // global_addr[31:0]
  g0[3] = (unsigned int)((ga >> 32) & 0x01FFFFFFu) | (2u << 30);  // addr|type=2
  v8u g1;
  g1[0] = (1u << 16);                                 // data_size=1 -> 2 bytes
  g1[1] = (elems_x & 0xFFFFu) << 16;                  // tensor_dim0 lo16
  g1[2] = ((elems_x >> 16) & 0xFFFFu) | ((tensor_rows & 0xFFFFu) << 16);
  g1[3] = ((tensor_rows >> 16) & 0xFFFFu) | ((elems_x & 0xFFFFu) << 16); // tile_dim0
  g1[4] = rows & 0xFFFFu;                             // tile_dim1 (tile_dim2=0)
  g1[5] = row_stride;                                 // tensor_dim0_stride lo32
  g1[6] = 0u;
  g1[7] = 0u;
  asm volatile("tensor_load_to_lds %0, %1" :: "s"(g0), "s"(g1) : "memory");
}

static __device__ __forceinline__ void tdm_wait() {
  asm volatile("s_wait_tensorcnt 0x0" ::: "memory");
}

// ---- prep kernels ----------------------------------------------------------

__global__ void k_convert_pad(const float* __restrict__ src, bf16_t* __restrict__ dst,
                              int M, int Ks, int Kd) {
  int i = blockIdx.x * blockDim.x + threadIdx.x;
  if (i >= M * Kd) return;
  int m = i / Kd, k = i - m * Kd;
  float v = (k < Ks) ? src[m * Ks + k] : 0.0f;
  dst[i] = (bf16_t)v;
}

static __device__ __forceinline__ float spatial_feat(int c, int n) {
  int y = n >> 6, x = n & 63;
  float gx = x * (1.0f / 63.0f), gy = y * (1.0f / 63.0f);
  switch (c) {
    case 0: return gx;
    case 1: return gy;
    case 2: return 1.0f - gx;
    case 3: return 1.0f - gy;
    default: return 0.5f * (gx + gy);
  }
}

__global__ void k_fill_mm(const float* __restrict__ images, const float* __restrict__ flows,
                          bf16_t* __restrict__ mm) {
  int i = blockIdx.x * blockDim.x + threadIdx.x;
  if (i >= BATCH * CPAD * SEQ) return;
  int n = i % SEQ; int t = i / SEQ; int c = t % CPAD; int b = t / CPAD;
  float v;
  if (c < 256)      v = images[((size_t)b * 256 + c) * SEQ + n];
  else if (c < 512) v = flows [((size_t)b * 256 + (c - 256)) * SEQ + n];
  else if (c < 520) v = spatial_feat(c - 512, n);
  else              v = 0.0f;
  mm[i] = (bf16_t)v;
}

__global__ void k_fill_fused_tail(bf16_t* __restrict__ fused) {
  int i = blockIdx.x * blockDim.x + threadIdx.x;
  if (i >= BATCH * 32 * SEQ) return;
  int n = i % SEQ; int t = i / SEQ; int j = t % 32; int b = t / 32;
  float v = (j < 8) ? spatial_feat(j, n) : 0.0f;
  fused[((size_t)b * CPAD + 512 + j) * SEQ + n] = (bf16_t)v;
}

// ---- generic bf16 WMMA GEMM: out[i][n] = bias[i] + sum_k A[i][k]*B[k][n] ---
// Double-buffered B tile; next chunk's global loads issued before the WMMA
// burst, LDS scatter after it, one barrier per iteration.
// mode 0: bf16 transposed [n][256]; mode 1: bf16 [i][4096]; mode 2: f32 [i][4096]

__launch_bounds__(256)
__global__ void k_gemm(const bf16_t* __restrict__ A, int lda,
                       const bf16_t* __restrict__ Bg, long long strideB,
                       const float* __restrict__ bias, int K,
                       void* __restrict__ Out, long long strideOut, int mode) {
  __shared__ bf16_t Blds[2][128][32];   // ping-pong, staged transposed [n][k]

  const int tid  = threadIdx.x;
  const int lane = tid & 31;
  const int wave = tid >> 5;
  const int half = lane >> 4;
  const int l16  = lane & 15;
  const int koff = half ? 8 : 0;

  const int iSub  = wave & 3;
  const int nHalf = wave >> 2;
  const int iBase = blockIdx.y * 64 + iSub * 16;
  const int nBlk  = blockIdx.x * 128;
  const int b     = blockIdx.z;

  const bf16_t* Bb   = Bg + (long long)b * strideB;
  const bf16_t* Arow = A + (long long)(iBase + l16) * lda;

  v8f acc[4] = {};

  // prologue: stage chunk 0 into buffer 0
  {
    unsigned long long st[4];
#pragma unroll
    for (int it = 0; it < 4; ++it) {
      int q = it * 256 + tid;
      int k = q >> 5, n4 = (q & 31) << 2;
      st[it] = *(const unsigned long long*)&Bb[(long long)k * SEQ + nBlk + n4];
    }
#pragma unroll
    for (int it = 0; it < 4; ++it) {
      int q = it * 256 + tid;
      int k = q >> 5, n4 = (q & 31) << 2;
      union { unsigned long long u; bf16_t h[4]; } d; d.u = st[it];
      Blds[0][n4 + 0][k] = d.h[0];
      Blds[0][n4 + 1][k] = d.h[1];
      Blds[0][n4 + 2][k] = d.h[2];
      Blds[0][n4 + 3][k] = d.h[3];
    }
  }
  __syncthreads();

  for (int k0 = 0; k0 < K; k0 += 32) {
    const int cur = (k0 >> 5) & 1, nxt = cur ^ 1;
    const int kn  = k0 + 32;
    const bool more = kn < K;

    // issue next chunk's global loads early (latency hidden by WMMAs below)
    unsigned long long st[4];
    if (more) {
#pragma unroll
      for (int it = 0; it < 4; ++it) {
        int q = it * 256 + tid;
        int k = q >> 5, n4 = (q & 31) << 2;
        st[it] = *(const unsigned long long*)&Bb[(long long)(kn + k) * SEQ + nBlk + n4];
      }
      __builtin_prefetch(Arow + kn, 0, 3);
    }

    v16bf a = load_frag(Arow + k0 + koff);
    v16bf bf0, bf1, bf2, bf3;
    {
      const bf16_t* bp0 = &Blds[cur][nHalf * 64 +  0 + l16][koff];
      const bf16_t* bp1 = &Blds[cur][nHalf * 64 + 16 + l16][koff];
      const bf16_t* bp2 = &Blds[cur][nHalf * 64 + 32 + l16][koff];
      const bf16_t* bp3 = &Blds[cur][nHalf * 64 + 48 + l16][koff];
      bf0 = make_frag(*(const uint4*)bp0, *(const uint4*)(bp0 + 16));
      bf1 = make_frag(*(const uint4*)bp1, *(const uint4*)(bp1 + 16));
      bf2 = make_frag(*(const uint4*)bp2, *(const uint4*)(bp2 + 16));
      bf3 = make_frag(*(const uint4*)bp3, *(const uint4*)(bp3 + 16));
    }
    acc[0] = __builtin_amdgcn_wmma_f32_16x16x32_bf16(false, a, false, bf0, (short)0, acc[0], false, false);
    acc[1] = __builtin_amdgcn_wmma_f32_16x16x32_bf16(false, a, false, bf1, (short)0, acc[1], false, false);
    acc[2] = __builtin_amdgcn_wmma_f32_16x16x32_bf16(false, a, false, bf2, (short)0, acc[2], false, false);
    acc[3] = __builtin_amdgcn_wmma_f32_16x16x32_bf16(false, a, false, bf3, (short)0, acc[3], false, false);

    // finish staging next chunk into the other buffer
    if (more) {
#pragma unroll
      for (int it = 0; it < 4; ++it) {
        int q = it * 256 + tid;
        int k = q >> 5, n4 = (q & 31) << 2;
        union { unsigned long long u; bf16_t h[4]; } d; d.u = st[it];
        Blds[nxt][n4 + 0][k] = d.h[0];
        Blds[nxt][n4 + 1][k] = d.h[1];
        Blds[nxt][n4 + 2][k] = d.h[2];
        Blds[nxt][n4 + 3][k] = d.h[3];
      }
    }
    __syncthreads();
  }

  const long long ob = (long long)b * strideOut;
#pragma unroll
  for (int t = 0; t < 4; ++t) {
#pragma unroll
    for (int r = 0; r < 8; ++r) {
      int i = iBase + r + 8 * half;
      int n = nBlk + nHalf * 64 + t * 16 + l16;
      float v = acc[t][r] + bias[i];
      if (mode == 0)      ((bf16_t*)Out)[ob + (long long)n * DIM + i] = (bf16_t)v;
      else if (mode == 1) ((bf16_t*)Out)[ob + (long long)i * SEQ + n] = (bf16_t)v;
      else                ((float*)Out)[ob + (long long)i * SEQ + n] = v;
    }
  }
}

// ---- flash attention (no scale, softmax over keys) -------------------------
// Q,K,V: [B][4096][256] bf16 row-major. Output: [B][256][4096] bf16.
// Double-buffered K/V tiles; K tile staged by the Tensor Data Mover (its
// 32x256 bf16 block is contiguous 16KB in global memory).

__launch_bounds__(256)
__global__ void k_flash(const bf16_t* __restrict__ Qg, const bf16_t* __restrict__ Kg,
                        const bf16_t* __restrict__ Vg, bf16_t* __restrict__ OutT) {
  __shared__ bf16_t Klds[2][32][256];   // key block, row-major
  __shared__ bf16_t Vlds[2][256][32];   // value block, transposed [a][m]
  __shared__ bf16_t Plds[8][16][32];    // per-wave P layout shuffle

  const int tid  = threadIdx.x;
  const int lane = tid & 31;
  const int wave = tid >> 5;
  const int half = lane >> 4;
  const int l16  = lane & 15;
  const int koff = half ? 8 : 0;
  const int b     = blockIdx.y;
  const int qBase = blockIdx.x * 128 + wave * 16;

  const bf16_t* Qb = Qg + (long long)b * SEQ * DIM;
  const bf16_t* Kb = Kg + (long long)b * SEQ * DIM;
  const bf16_t* Vb = Vg + (long long)b * SEQ * DIM;

  // Q fragments for this wave's 16 rows, kept resident (8 x 32-K chunks)
  v16bf qf[8];
  const bf16_t* Qrow = Qb + (long long)(qBase + l16) * DIM;
#pragma unroll
  for (int kc = 0; kc < 8; ++kc) qf[kc] = load_frag(Qrow + kc * 32 + koff);

  v8f acc[16] = {};
  float mrun[8], lrun[8];
#pragma unroll
  for (int r = 0; r < 8; ++r) { mrun[r] = -1e30f; lrun[r] = 0.0f; }

  const int vr  = tid >> 3;
  const int va0 = (tid & 7) * 32;

  const uint32_t kOff0 = (uint32_t)(uintptr_t)(lds_void_t*)&Klds[0][0][0];
  const uint32_t kOff1 = (uint32_t)(uintptr_t)(lds_void_t*)&Klds[1][0][0];

  // prologue: stage tile 0 into buffer 0
  {
    if (wave == 0) tdm_load_2d(kOff0, Kb, DIM, 32, DIM, SEQ);
    const bf16_t* src = &Vb[(long long)vr * DIM + va0];
    uint4 vv0[4];
#pragma unroll
    for (int it = 0; it < 4; ++it) vv0[it] = *(const uint4*)(src + it * 8);
#pragma unroll
    for (int it = 0; it < 4; ++it) {
      union { uint4 u; bf16_t h[8]; } d; d.u = vv0[it];
#pragma unroll
      for (int j = 0; j < 8; ++j) Vlds[0][va0 + it * 8 + j][vr] = d.h[j];
    }
    tdm_wait();
  }
  __syncthreads();

  for (int m0 = 0; m0 < SEQ; m0 += 32) {
    const int cur = (m0 >> 5) & 1, nxt = cur ^ 1;
    const int mn  = m0 + 32;
    const bool more = mn < SEQ;

    // kick off next tile's staging (overlaps with compute below)
    uint4 vv[4];
    if (more) {
      if (wave == 0)
        tdm_load_2d(nxt ? kOff1 : kOff0, Kb + (long long)mn * DIM, DIM, 32, DIM, SEQ - mn);
      const bf16_t* src = &Vb[(long long)(mn + vr) * DIM + va0];
#pragma unroll
      for (int it = 0; it < 4; ++it) vv[it] = *(const uint4*)(src + it * 8);
    }

    // S(16x32) = Q(16x256) * K_blk^T
    v8f s0 = {}, s1 = {};
#pragma unroll
    for (int kc = 0; kc < 8; ++kc) {
      const bf16_t* b0 = &Klds[cur][l16][kc * 32 + koff];
      const bf16_t* b1 = &Klds[cur][16 + l16][kc * 32 + koff];
      v16bf kf0 = make_frag(*(const uint4*)b0, *(const uint4*)(b0 + 16));
      v16bf kf1 = make_frag(*(const uint4*)b1, *(const uint4*)(b1 + 16));
      s0 = __builtin_amdgcn_wmma_f32_16x16x32_bf16(false, qf[kc], false, kf0,
                                                   (short)0, s0, false, false);
      s1 = __builtin_amdgcn_wmma_f32_16x16x32_bf16(false, qf[kc], false, kf1,
                                                   (short)0, s1, false, false);
    }

    // online softmax
    float p0[8], p1[8], scale[8];
#pragma unroll
    for (int r = 0; r < 8; ++r) {
      float v = fmaxf(s0[r], s1[r]);
#pragma unroll
      for (int off = 8; off >= 1; off >>= 1) v = fmaxf(v, __shfl_xor(v, off, 16));
      float nm = fmaxf(mrun[r], v);
      scale[r] = __expf(mrun[r] - nm);
      mrun[r]  = nm;
      p0[r] = __expf(s0[r] - nm);
      p1[r] = __expf(s1[r] - nm);
      float rs = p0[r] + p1[r];
#pragma unroll
      for (int off = 8; off >= 1; off >>= 1) rs += __shfl_xor(rs, off, 16);
      lrun[r] = lrun[r] * scale[r] + rs;
    }
#pragma unroll
    for (int t = 0; t < 16; ++t)
#pragma unroll
      for (int r = 0; r < 8; ++r) acc[t][r] *= scale[r];

    // C-layout -> A-layout for P via per-wave LDS slice (wave-local DS order)
#pragma unroll
    for (int r = 0; r < 8; ++r) {
      Plds[wave][r + 8 * half][l16]      = (bf16_t)p0[r];
      Plds[wave][r + 8 * half][16 + l16] = (bf16_t)p1[r];
    }
    const bf16_t* pp = &Plds[wave][l16][koff];
    v16bf pf = make_frag(*(const uint4*)pp, *(const uint4*)(pp + 16));

    // O(16x256) += P(16x32) * V_blk(32x256)
#pragma unroll
    for (int t = 0; t < 16; t += 4) {
      const bf16_t* vp0 = &Vlds[cur][(t + 0) * 16 + l16][koff];
      const bf16_t* vp1 = &Vlds[cur][(t + 1) * 16 + l16][koff];
      const bf16_t* vp2 = &Vlds[cur][(t + 2) * 16 + l16][koff];
      const bf16_t* vp3 = &Vlds[cur][(t + 3) * 16 + l16][koff];
      v16bf vf0 = make_frag(*(const uint4*)vp0, *(const uint4*)(vp0 + 16));
      v16bf vf1 = make_frag(*(const uint4*)vp1, *(const uint4*)(vp1 + 16));
      v16bf vf2 = make_frag(*(const uint4*)vp2, *(const uint4*)(vp2 + 16));
      v16bf vf3 = make_frag(*(const uint4*)vp3, *(const uint4*)(vp3 + 16));
      acc[t + 0] = __builtin_amdgcn_wmma_f32_16x16x32_bf16(false, pf, false, vf0, (short)0, acc[t + 0], false, false);
      acc[t + 1] = __builtin_amdgcn_wmma_f32_16x16x32_bf16(false, pf, false, vf1, (short)0, acc[t + 1], false, false);
      acc[t + 2] = __builtin_amdgcn_wmma_f32_16x16x32_bf16(false, pf, false, vf2, (short)0, acc[t + 2], false, false);
      acc[t + 3] = __builtin_amdgcn_wmma_f32_16x16x32_bf16(false, pf, false, vf3, (short)0, acc[t + 3], false, false);
    }

    // finish staging next tile into the other buffers
    if (more) {
#pragma unroll
      for (int it = 0; it < 4; ++it) {
        union { uint4 u; bf16_t h[8]; } d; d.u = vv[it];
#pragma unroll
        for (int j = 0; j < 8; ++j) Vlds[nxt][va0 + it * 8 + j][vr] = d.h[j];
      }
    }
    tdm_wait();
    __syncthreads();
  }

  // normalize and store transposed: OutT[b][a][n]
  bf16_t* Ob = OutT + (long long)b * DIM * SEQ;
#pragma unroll
  for (int r = 0; r < 8; ++r) {
    float inv = 1.0f / lrun[r];
    int n = qBase + r + 8 * half;
#pragma unroll
    for (int t = 0; t < 16; ++t) {
      int a = t * 16 + l16;
      Ob[(long long)a * SEQ + n] = (bf16_t)(acc[t][r] * inv);
    }
  }
}

// ---- host orchestration ----------------------------------------------------

extern "C" void kernel_launch(void* const* d_in, const int* in_sizes, int n_in,
                              void* d_out, int out_size, void* d_ws, size_t ws_size,
                              hipStream_t stream) {
  (void)in_sizes; (void)n_in; (void)out_size; (void)ws_size;

  const float* images  = (const float*)d_in[0];
  const float* flows   = (const float*)d_in[1];
  const float* img_wt  = (const float*)d_in[2];
  const float* img_bt  = (const float*)d_in[3];
  const float* img_wp  = (const float*)d_in[4];
  const float* img_bp  = (const float*)d_in[5];
  const float* img_wv  = (const float*)d_in[6];
  const float* img_bv  = (const float*)d_in[7];
  const float* img_wo  = (const float*)d_in[8];
  const float* img_bo  = (const float*)d_in[9];
  const float* lang_wt = (const float*)d_in[10];
  const float* lang_bt = (const float*)d_in[11];
  const float* lang_wp = (const float*)d_in[12];
  const float* lang_bp = (const float*)d_in[13];
  const float* lang_wv = (const float*)d_in[14];
  const float* lang_bv = (const float*)d_in[15];
  const float* lang_wo = (const float*)d_in[16];
  const float* lang_bo = (const float*)d_in[17];
  const float* fus_w   = (const float*)d_in[18];
  const float* fus_b   = (const float*)d_in[19];

  char* ws = (char*)d_ws;
  size_t off = 0;
  auto alloc = [&](size_t bytes) -> void* {
    void* p = ws + off;
    off = (off + bytes + 255) & ~(size_t)255;
    return p;
  };

  const size_t szMM = (size_t)BATCH * CPAD * SEQ * sizeof(bf16_t);
  const size_t szNE = (size_t)BATCH * SEQ * DIM * sizeof(bf16_t);

  bf16_t* mm    = (bf16_t*)alloc(szMM);
  bf16_t* Qi    = (bf16_t*)alloc(szNE);
  bf16_t* Ki    = (bf16_t*)alloc(szNE);
  bf16_t* Vi    = (bf16_t*)alloc(szNE);
  bf16_t* Ql    = (bf16_t*)alloc(szNE);
  bf16_t* Kl    = (bf16_t*)alloc(szNE);
  bf16_t* Vl    = (bf16_t*)alloc(szNE);
  bf16_t* attI  = (bf16_t*)alloc(szNE);
  bf16_t* attL  = (bf16_t*)alloc(szNE);
  bf16_t* fused = (bf16_t*)alloc(szMM);
  bf16_t* wIT   = (bf16_t*)alloc((size_t)256 * CPAD * 2);
  bf16_t* wIP   = (bf16_t*)alloc((size_t)256 * CPAD * 2);
  bf16_t* wIV   = (bf16_t*)alloc((size_t)256 * CPAD * 2);
  bf16_t* wLT   = (bf16_t*)alloc((size_t)256 * 256 * 2);
  bf16_t* wLP   = (bf16_t*)alloc((size_t)256 * 256 * 2);
  bf16_t* wLV   = (bf16_t*)alloc((size_t)256 * 256 * 2);
  bf16_t* wOI   = (bf16_t*)alloc((size_t)256 * 256 * 2);
  bf16_t* wOL   = (bf16_t*)alloc((size_t)256 * 256 * 2);
  bf16_t* wF    = (bf16_t*)alloc((size_t)256 * CPAD * 2);

  const int thr = 256;
  const int g544 = (256 * CPAD + thr - 1) / thr;
  const int g256 = (256 * 256 + thr - 1) / thr;

  k_convert_pad<<<g544, thr, 0, stream>>>(img_wt,  wIT, 256, 520, CPAD);
  k_convert_pad<<<g544, thr, 0, stream>>>(img_wp,  wIP, 256, 520, CPAD);
  k_convert_pad<<<g544, thr, 0, stream>>>(img_wv,  wIV, 256, 520, CPAD);
  k_convert_pad<<<g256, thr, 0, stream>>>(lang_wt, wLT, 256, 256, 256);
  k_convert_pad<<<g256, thr, 0, stream>>>(lang_wp, wLP, 256, 256, 256);
  k_convert_pad<<<g256, thr, 0, stream>>>(lang_wv, wLV, 256, 256, 256);
  k_convert_pad<<<g256, thr, 0, stream>>>(img_wo,  wOI, 256, 256, 256);
  k_convert_pad<<<g256, thr, 0, stream>>>(lang_wo, wOL, 256, 256, 256);
  k_convert_pad<<<g544, thr, 0, stream>>>(fus_w,   wF,  256, 520, CPAD);

  k_fill_mm<<<(BATCH * CPAD * SEQ + thr - 1) / thr, thr, 0, stream>>>(images, flows, mm);
  k_fill_fused_tail<<<(BATCH * 32 * SEQ + thr - 1) / thr, thr, 0, stream>>>(fused);

  const long long sMM = (long long)CPAD * SEQ;
  const long long sQ  = (long long)SEQ * DIM;
  dim3 gg(SEQ / 128, 256 / 64, BATCH);

  k_gemm<<<gg, thr, 0, stream>>>(wIT, CPAD, mm, sMM, img_bt, CPAD, Qi, sQ, 0);
  k_gemm<<<gg, thr, 0, stream>>>(wIP, CPAD, mm, sMM, img_bp, CPAD, Ki, sQ, 0);
  k_gemm<<<gg, thr, 0, stream>>>(wIV, CPAD, mm, sMM, img_bv, CPAD, Vi, sQ, 0);
  const bf16_t* mmFlows = mm + (size_t)256 * SEQ;
  k_gemm<<<gg, thr, 0, stream>>>(wLT, 256, mmFlows, sMM, lang_bt, 256, Ql, sQ, 0);
  k_gemm<<<gg, thr, 0, stream>>>(wLP, 256, mmFlows, sMM, lang_bp, 256, Kl, sQ, 0);
  k_gemm<<<gg, thr, 0, stream>>>(wLV, 256, mmFlows, sMM, lang_bv, 256, Vl, sQ, 0);

  dim3 gf(SEQ / 128, BATCH);
  k_flash<<<gf, thr, 0, stream>>>(Qi, Ki, Vi, attI);
  k_flash<<<gf, thr, 0, stream>>>(Ql, Kl, Vl, attL);

  k_gemm<<<gg, thr, 0, stream>>>(wOI, 256, attI, sQ, img_bo, 256, fused, sMM, 1);
  k_gemm<<<gg, thr, 0, stream>>>(wOL, 256, attL, sQ, lang_bo, 256,
                                 fused + (size_t)256 * SEQ, sMM, 1);

  k_gemm<<<gg, thr, 0, stream>>>(wF, CPAD, fused, sMM, fus_b, CPAD,
                                 d_out, (long long)256 * SEQ, 2);
}
